// SelfAttention_22943715295487
// MI455X (gfx1250) — matrix operand order repaired and checked by reading, not compiled, not run on hardware
//
#include <hip/hip_runtime.h>
#include <stdint.h>

// ---------------- types ----------------
typedef __attribute__((ext_vector_type(16))) __bf16    v16bf;
typedef __attribute__((ext_vector_type(8)))  float     v8f;
typedef __attribute__((ext_vector_type(4)))  uint32_t  u32x4;
typedef __attribute__((ext_vector_type(2)))  uint32_t  u32x2;
typedef __attribute__((ext_vector_type(4)))  float     f32x4;

#define DIM   1024
#define NHEAD 16
#define HDIM  64
#define SEQ   2048
#define BATCH 2
#define MTOT  (BATCH * SEQ)   // 4096 rows total

__device__ __forceinline__ uint16_t f2bf(float f) {
  uint32_t u = __builtin_bit_cast(uint32_t, f);
  u += 0x7FFFu + ((u >> 16) & 1u);            // round-to-nearest-even
  return (uint16_t)(u >> 16);
}
__device__ __forceinline__ float bf2f(uint16_t h) {
  uint32_t u = ((uint32_t)h) << 16;
  return __builtin_bit_cast(float, u);
}

union FragBF { u32x4 u4[2]; v16bf v; };       // 32 bytes = one A/B fragment
union Pack8  { u32x4 q; uint16_t h[8]; };

// ---- CDNA5 async global->LDS DMA (ASYNCcnt-tracked, ISA §15.18.3 op 98) ----
// For a generic pointer into shared memory, addr[31:0] is the LDS byte address
// (ISA §10.2 aperture rules), so the VDST operand is the truncated pointer.
__device__ __forceinline__ void async_load_b128(void* lds_dst, const void* gsrc) {
  uint32_t loff = (uint32_t)(uintptr_t)lds_dst;
  asm volatile("global_load_async_to_lds_b128 %0, %1, off"
               :: "v"(loff), "v"(gsrc) : "memory");
}
__device__ __forceinline__ void wait_async0() {
  asm volatile("s_wait_asynccnt 0" ::: "memory");
}

// ---------------- fp32 -> bf16 convert ----------------
__global__ __launch_bounds__(256) void convert_bf16_kernel(
    const float* __restrict__ in, uint16_t* __restrict__ out, int n) {
  int i = (blockIdx.x * 256 + threadIdx.x) * 4;
  if (i >= n) return;
  f32x4 f = *(const f32x4*)(in + i);
  u32x2 packed;
  packed.x = (uint32_t)f2bf(f[0]) | ((uint32_t)f2bf(f[1]) << 16);
  packed.y = (uint32_t)f2bf(f[2]) | ((uint32_t)f2bf(f[3]) << 16);
  *(u32x2*)(out + i) = packed;
}

// ---------------- tiled WMMA GEMM:  C[M,N] = A[M,K] @ Bw[N,K]^T + bias ----------------
// block tile 128x128, 256 threads = 8 waves (4 x Mdim, 2 x Ndim), wave tile 32x64.
// Double-buffered LDS, staged by async DMA: prefetch tile k+1 while WMMAs eat tile k.
__global__ __launch_bounds__(256) void gemm_bf16_wmma(
    const uint16_t* __restrict__ A,    // [M,K] bf16
    const uint16_t* __restrict__ Bw,   // [N,K] bf16 (row n = weight row n)
    const float*    __restrict__ bias, // [N]
    uint16_t* __restrict__ Cbf, float* __restrict__ Cf,
    int M, int N, int K, int out_bf16)
{
  __shared__ __align__(16) uint16_t lds_a[2][128 * 32];
  __shared__ __align__(16) uint16_t lds_b[2][128 * 32];
  const int tid  = threadIdx.x;
  const int lane = tid & 31;
  const int w    = tid >> 5;
  const int wm   = w & 3;            // M sub-tile (x32)
  const int wn   = w >> 2;           // N sub-tile (x64)
  const int m_block = blockIdx.y * 128;
  const int n_block = blockIdx.x * 128;
  const int half = (lane >= 16) ? 1 : 0;
  const int l16  = lane & 15;

  v8f acc[2][4];
  #pragma unroll
  for (int mt = 0; mt < 2; mt++)
    #pragma unroll
    for (int nt = 0; nt < 4; nt++)
      #pragma unroll
      for (int g = 0; g < 8; g++) acc[mt][nt][g] = 0.f;

  // async-stage one 128x32 A tile + 128x32 B tile into LDS buffer `buf`
  auto stage = [&](int buf, int k0) {
    #pragma unroll
    for (int i = 0; i < 2; i++) {
      int c    = tid + 256 * i;          // 0..511 chunks of 8 bf16
      int row  = c >> 2;
      int col8 = (c & 3) * 8;
      async_load_b128(&lds_a[buf][row * 32 + col8],
                      &A[(size_t)(m_block + row) * K + k0 + col8]);
      async_load_b128(&lds_b[buf][row * 32 + col8],
                      &Bw[(size_t)(n_block + row) * K + k0 + col8]);
    }
  };

  const int nsteps = K / 32;
  stage(0, 0);
  wait_async0();
  __syncthreads();

  for (int ks = 0; ks < nsteps; ks++) {
    const int cur = ks & 1;
    if (ks + 1 < nsteps) stage(cur ^ 1, (ks + 1) * 32);   // prefetch next tile

    // A fragments: lane holds row M=l16; K chunks at (half*8) and 16+(half*8)
    FragBF af[2];
    #pragma unroll
    for (int mt = 0; mt < 2; mt++) {
      int row = wm * 32 + mt * 16 + l16;
      int c0  = half * 8;
      af[mt].u4[0] = *(const u32x4*)&lds_a[cur][row * 32 + c0];
      af[mt].u4[1] = *(const u32x4*)&lds_a[cur][row * 32 + c0 + 16];
    }
    // B fragments: lane holds column N=l16 = weight row; K chunks at (half*8) and 16+(half*8)
    FragBF bfr[4];
    #pragma unroll
    for (int nt = 0; nt < 4; nt++) {
      int nrow = wn * 64 + nt * 16 + l16;
      int koff = half * 8;
      bfr[nt].u4[0] = *(const u32x4*)&lds_b[cur][nrow * 32 + koff];
      bfr[nt].u4[1] = *(const u32x4*)&lds_b[cur][nrow * 32 + koff + 16];
    }
    #pragma unroll
    for (int mt = 0; mt < 2; mt++)
      #pragma unroll
      for (int nt = 0; nt < 4; nt++)
        acc[mt][nt] = __builtin_amdgcn_wmma_f32_16x16x32_bf16(
            false, af[mt].v, false, bfr[nt].v, (short)0, acc[mt][nt], false, false);

    wait_async0();      // next buffer fully landed in LDS
    __syncthreads();    // all waves done reading `cur`, prefetch visible to all
  }

  // epilogue: C layout — VGPR g: row g (lanes 0-15) / row g+8 (lanes 16-31), col = l16
  #pragma unroll
  for (int mt = 0; mt < 2; mt++)
    #pragma unroll
    for (int nt = 0; nt < 4; nt++)
      #pragma unroll
      for (int g = 0; g < 8; g++) {
        int row = m_block + wm * 32 + mt * 16 + g + half * 8;
        int col = n_block + wn * 64 + nt * 16 + l16;
        float val = acc[mt][nt][g] + bias[col];
        if (out_bf16) Cbf[(size_t)row * N + col] = f2bf(val);
        else          Cf [(size_t)row * N + col] = val;
      }
}

// ---------------- RMSNorm in-place over bf16 rows ----------------
__global__ __launch_bounds__(256) void rmsnorm_bf16_kernel(
    uint16_t* __restrict__ buf, const float* __restrict__ w, int ncols) {
  const int row = blockIdx.x;
  const int tid = threadIdx.x;
  uint16_t* p = buf + (size_t)row * ncols;
  float vals[4];
  float ss = 0.f;
  #pragma unroll
  for (int i = 0; i < 4; i++) {
    int c = tid + i * 256;
    vals[i] = bf2f(p[c]);
    ss += vals[i] * vals[i];
  }
  #pragma unroll
  for (int off = 16; off > 0; off >>= 1) ss += __shfl_xor(ss, off, 32);
  __shared__ float red[8];
  if ((tid & 31) == 0) red[tid >> 5] = ss;
  __syncthreads();
  float tot = 0.f;
  #pragma unroll
  for (int i = 0; i < 8; i++) tot += red[i];
  float sc = rsqrtf(tot / (float)ncols + 1e-6f);
  #pragma unroll
  for (int i = 0; i < 4; i++) {
    int c = tid + i * 256;
    p[c] = f2bf(vals[i] * sc * w[c]);
  }
}

// ---------------- flash attention: one block per (qblock64, head, batch) ----------------
// 128 threads = 4 waves; wave owns a 16-row Q tile. KV processed in 64-row blocks.
// K tile staged by async DMA; V tile staged transposed through VGPRs.
__global__ __launch_bounds__(128) void attention_wmma(
    const uint16_t* __restrict__ Q, const uint16_t* __restrict__ Kb,
    const uint16_t* __restrict__ Vb, uint16_t* __restrict__ Ob)
{
  __shared__ __align__(16) uint16_t lds_k [64 * 64];       // K block, row-major [kv][d]
  __shared__ __align__(16) uint16_t lds_vt[64 * 64];       // V block transposed [d][kv]
  __shared__ __align__(16) uint16_t lds_p [4 * 16 * 64];   // per-wave P staging

  const int tid  = threadIdx.x;
  const int lane = tid & 31;
  const int w    = tid >> 5;
  const int half = (lane >= 16) ? 1 : 0;
  const int l16  = lane & 15;
  const int h    = blockIdx.y;
  const int b    = blockIdx.z;
  const size_t rowbase = (size_t)b * SEQ;
  const int col0 = h * HDIM;
  const int q0   = blockIdx.x * 64 + w * 16;
  const float scale = 0.125f;                              // 1/sqrt(64)

  // Q A-fragments, kept in registers (HDIM=64 -> two K=32 steps)
  FragBF qf[2];
  {
    const uint16_t* qp = Q + (rowbase + q0 + l16) * DIM + col0;
    #pragma unroll
    for (int ks = 0; ks < 2; ks++) {
      int c0 = ks * 32 + half * 8;
      qf[ks].u4[0] = *(const u32x4*)&qp[c0];
      qf[ks].u4[1] = *(const u32x4*)&qp[c0 + 16];
    }
  }

  v8f Oacc[4];
  float mrow[8], lrow[8];
  #pragma unroll
  for (int dt = 0; dt < 4; dt++)
    #pragma unroll
    for (int g = 0; g < 8; g++) Oacc[dt][g] = 0.f;
  #pragma unroll
  for (int g = 0; g < 8; g++) { mrow[g] = -1e30f; lrow[g] = 0.f; }

  for (int kv0 = 0; kv0 < SEQ; kv0 += 64) {
    __syncthreads();
    // stage K via async DMA; V via VGPR transpose: 512 chunks of 8 bf16, 4/thread
    #pragma unroll
    for (int i = 0; i < 4; i++) {
      int c  = tid + 128 * i;
      int r  = c >> 3;
      int c8 = (c & 7) * 8;
      async_load_b128(&lds_k[r * 64 + c8],
                      &Kb[(rowbase + kv0 + r) * DIM + col0 + c8]);
      Pack8 pv;
      pv.q = *(const u32x4*)&Vb[(rowbase + kv0 + r) * DIM + col0 + c8];
      #pragma unroll
      for (int j = 0; j < 8; j++) lds_vt[(c8 + j) * 64 + r] = pv.h[j];
    }
    wait_async0();
    __syncthreads();

    // scores S = Q K^T : 4 kv-tiles x 2 k-steps = 8 WMMAs
    v8f sc[4];
    #pragma unroll
    for (int nt = 0; nt < 4; nt++) {
      v8f s = {0.f, 0.f, 0.f, 0.f, 0.f, 0.f, 0.f, 0.f};
      #pragma unroll
      for (int ks = 0; ks < 2; ks++) {
        FragBF kf;
        int kvr  = nt * 16 + l16;
        int koff = ks * 32 + half * 8;
        kf.u4[0] = *(const u32x4*)&lds_k[kvr * 64 + koff];
        kf.u4[1] = *(const u32x4*)&lds_k[kvr * 64 + koff + 16];
        s = __builtin_amdgcn_wmma_f32_16x16x32_bf16(
            false, qf[ks].v, false, kf.v, (short)0, s, false, false);
      }
      sc[nt] = s;
    }

    // online softmax; row lives across 16 lanes in one acc VGPR
    #pragma unroll
    for (int g = 0; g < 8; g++) {
      float s0 = sc[0][g] * scale, s1 = sc[1][g] * scale;
      float s2 = sc[2][g] * scale, s3 = sc[3][g] * scale;
      float vm = fmaxf(fmaxf(s0, s1), fmaxf(s2, s3));
      #pragma unroll
      for (int off = 8; off > 0; off >>= 1) vm = fmaxf(vm, __shfl_xor(vm, off, 16));
      float mnew  = fmaxf(mrow[g], vm);
      float alpha = __expf(mrow[g] - mnew);
      float p0 = __expf(s0 - mnew), p1 = __expf(s1 - mnew);
      float p2 = __expf(s2 - mnew), p3 = __expf(s3 - mnew);
      float rs = p0 + p1 + p2 + p3;
      #pragma unroll
      for (int off = 8; off > 0; off >>= 1) rs += __shfl_xor(rs, off, 16);
      lrow[g] = lrow[g] * alpha + rs;
      mrow[g] = mnew;
      #pragma unroll
      for (int dt = 0; dt < 4; dt++) Oacc[dt][g] *= alpha;
      // stash P (C-layout -> LDS) for A-fragment reload
      uint16_t* pl = &lds_p[(w * 16 + g + half * 8) * 64];
      pl[ 0 + l16] = f2bf(p0);
      pl[16 + l16] = f2bf(p1);
      pl[32 + l16] = f2bf(p2);
      pl[48 + l16] = f2bf(p3);
    }
    __syncthreads();   // also covers same-wave LDS P write->read

    // O += P @ V : 2 kv-steps x 4 d-tiles = 8 WMMAs
    #pragma unroll
    for (int ks = 0; ks < 2; ks++) {
      FragBF pf;
      {
        const uint16_t* pl = &lds_p[(w * 16 + l16) * 64];
        int c0 = ks * 32 + half * 8;
        pf.u4[0] = *(const u32x4*)&pl[c0];
        pf.u4[1] = *(const u32x4*)&pl[c0 + 16];
      }
      #pragma unroll
      for (int dt = 0; dt < 4; dt++) {
        FragBF vf;
        int d     = dt * 16 + l16;
        int kvoff = ks * 32 + half * 8;
        vf.u4[0] = *(const u32x4*)&lds_vt[d * 64 + kvoff];
        vf.u4[1] = *(const u32x4*)&lds_vt[d * 64 + kvoff + 16];
        Oacc[dt] = __builtin_amdgcn_wmma_f32_16x16x32_bf16(
            false, pf.v, false, vf.v, (short)0, Oacc[dt], false, false);
      }
    }
  }

  // final 1/l scaling, write bf16 attention output
  #pragma unroll
  for (int dt = 0; dt < 4; dt++)
    #pragma unroll
    for (int g = 0; g < 8; g++) {
      int row = q0 + g + half * 8;
      int col = col0 + dt * 16 + l16;
      Ob[(rowbase + row) * DIM + col] = f2bf(Oacc[dt][g] / lrow[g]);
    }
}

// ---------------- launch ----------------
extern "C" void kernel_launch(void* const* d_in, const int* in_sizes, int n_in,
                              void* d_out, int out_size, void* d_ws, size_t ws_size,
                              hipStream_t stream) {
  (void)in_sizes; (void)n_in; (void)out_size; (void)ws_size;
  const float* x   = (const float*)d_in[0];
  const float* Wq  = (const float*)d_in[1];
  const float* bq  = (const float*)d_in[2];
  const float* Wk  = (const float*)d_in[3];
  const float* bk  = (const float*)d_in[4];
  const float* Wv  = (const float*)d_in[5];
  const float* bv  = (const float*)d_in[6];
  const float* Wo  = (const float*)d_in[7];
  const float* bo  = (const float*)d_in[8];
  const float* nqw = (const float*)d_in[9];
  const float* nkw = (const float*)d_in[10];
  float* out = (float*)d_out;

  char* ws = (char*)d_ws;
  size_t off = 0;
  auto alloc = [&](size_t bytes) -> void* {
    void* p = ws + off;
    off += (bytes + 255) & ~(size_t)255;
    return p;
  };
  const size_t actB = (size_t)MTOT * DIM * 2;   // 8 MB
  const size_t wB   = (size_t)DIM * DIM * 2;    // 2 MB
  uint16_t* xb  = (uint16_t*)alloc(actB);
  uint16_t* Wqb = (uint16_t*)alloc(wB);
  uint16_t* Wkb = (uint16_t*)alloc(wB);
  uint16_t* Wvb = (uint16_t*)alloc(wB);
  uint16_t* Wob = (uint16_t*)alloc(wB);
  uint16_t* qb  = (uint16_t*)alloc(actB);
  uint16_t* kb  = (uint16_t*)alloc(actB);
  uint16_t* vb  = (uint16_t*)alloc(actB);
  uint16_t* ab  = (uint16_t*)alloc(actB);       // total ~48 MB

  // 1) fp32 -> bf16 conversions
  {
    int n = MTOT * DIM;
    convert_bf16_kernel<<<n / 1024, 256, 0, stream>>>(x, xb, n);
    int nw = DIM * DIM;
    convert_bf16_kernel<<<nw / 1024, 256, 0, stream>>>(Wq, Wqb, nw);
    convert_bf16_kernel<<<nw / 1024, 256, 0, stream>>>(Wk, Wkb, nw);
    convert_bf16_kernel<<<nw / 1024, 256, 0, stream>>>(Wv, Wvb, nw);
    convert_bf16_kernel<<<nw / 1024, 256, 0, stream>>>(Wo, Wob, nw);
  }

  // 2) Q/K/V projections (x @ W^T + b), bf16 out
  dim3 ggrid(DIM / 128, MTOT / 128);
  gemm_bf16_wmma<<<ggrid, 256, 0, stream>>>(xb, Wqb, bq, qb, nullptr, MTOT, DIM, DIM, 1);
  gemm_bf16_wmma<<<ggrid, 256, 0, stream>>>(xb, Wkb, bk, kb, nullptr, MTOT, DIM, DIM, 1);
  gemm_bf16_wmma<<<ggrid, 256, 0, stream>>>(xb, Wvb, bv, vb, nullptr, MTOT, DIM, DIM, 1);

  // 3) RMSNorm over full DIM on q and k (pre head-split, as in reference)
  rmsnorm_bf16_kernel<<<MTOT, 256, 0, stream>>>(qb, nqw, DIM);
  rmsnorm_bf16_kernel<<<MTOT, 256, 0, stream>>>(kb, nkw, DIM);

  // 4) attention
  dim3 agrid(SEQ / 64, NHEAD, BATCH);
  attention_wmma<<<agrid, 128, 0, stream>>>(qb, kb, vb, ab);

  // 5) output projection, f32 out + bias
  gemm_bf16_wmma<<<ggrid, 256, 0, stream>>>(ab, Wob, bo, nullptr, out, MTOT, DIM, DIM, 0);
}
